// EfficientGlobalPointer_21337397527383
// MI455X (gfx1250) — matrix operand order repaired and checked by reading, not compiled, not run on hardware
//
#include <hip/hip_runtime.h>
#include <hip/hip_bf16.h>

// Problem constants
#define Bsz 8
#define Lsz 1024
#define Hsz 768
#define Dsz 64
#define NHEADS 12

typedef __attribute__((ext_vector_type(16))) _Float16 v16h;
typedef __attribute__((ext_vector_type(8)))  float    v8f;
typedef __attribute__((ext_vector_type(4)))  float    f32x4;
typedef __attribute__((ext_vector_type(4)))  int      i32x4;

struct alignas(16) H8 { _Float16 h[8]; };

// ---------------------------------------------------------------------------
// Kernel 0: fold W2 through W1:  W12 = W1 @ W2  (768x24),  bfull = b1@W2 + b2
// ---------------------------------------------------------------------------
__global__ void __launch_bounds__(256) w12_kernel(
    const float* __restrict__ W1, const float* __restrict__ b1,
    const float* __restrict__ W2, const float* __restrict__ b2,
    float* __restrict__ W12, float* __restrict__ bfull)
{
    const int idx = blockIdx.x * 256 + threadIdx.x;
    if (idx < Hsz * 2 * NHEADS) {
        const int k = idx / (2 * NHEADS);
        const int j = idx % (2 * NHEADS);
        float s = 0.0f;
        for (int c = 0; c < 2 * Dsz; ++c)
            s = fmaf(W1[k * (2 * Dsz) + c], W2[c * (2 * NHEADS) + j], s);
        W12[idx] = s;
    }
    if (blockIdx.x == 0 && threadIdx.x < 2 * NHEADS) {
        const int j = threadIdx.x;
        float s = b2[j];
        for (int c = 0; c < 2 * Dsz; ++c)
            s = fmaf(b1[c], W2[c * (2 * NHEADS) + j], s);
        bfull[j] = s;
    }
}

// ---------------------------------------------------------------------------
// Kernel 1: seq = hidden @ W1 + b1 (cols 0..127) and bias = hidden@W12 + bfull
// (cols 128..151).  RoPE applied to seq; q/k stored as f16 for WMMA.
// Bias stored as separate planes qbias/kbias[(b*12+h)*1024 + pos] so kernel 2
// can do vectorized loads in row-major output layout.
// ---------------------------------------------------------------------------
__global__ void __launch_bounds__(256) seq_rope_kernel(
    const float* __restrict__ hidden,
    const float* __restrict__ W1,  const float* __restrict__ b1,
    const float* __restrict__ W12, const float* __restrict__ bfull,
    _Float16* __restrict__ qws, _Float16* __restrict__ kws,
    float* __restrict__ qbias, float* __restrict__ kbias)
{
    __shared__ float lds[16 * Hsz];
    const int tid     = threadIdx.x;
    const int rowbase = blockIdx.x * 16;          // 0..8191 step 16

    // cooperative load of 16x768 f32 tile (coalesced)
    const float* src = hidden + (size_t)rowbase * Hsz;
    for (int i = tid; i < 16 * Hsz; i += 256) lds[i] = src[i];
    __syncthreads();

    if (tid < 152) {
        float acc[16];
        const float init = (tid < 128) ? b1[tid] : bfull[tid - 128];
        #pragma unroll
        for (int r = 0; r < 16; ++r) acc[r] = init;

        for (int k = 0; k < Hsz; k += 4) {
            float w0, w1, w2, w3;
            if (tid < 128) {
                w0 = W1[(k + 0) * 128 + tid];
                w1 = W1[(k + 1) * 128 + tid];
                w2 = W1[(k + 2) * 128 + tid];
                w3 = W1[(k + 3) * 128 + tid];
            } else {
                const int j = tid - 128;
                w0 = W12[(k + 0) * 24 + j];
                w1 = W12[(k + 1) * 24 + j];
                w2 = W12[(k + 2) * 24 + j];
                w3 = W12[(k + 3) * 24 + j];
            }
            #pragma unroll
            for (int r = 0; r < 16; ++r) {
                const float4 x = *(const float4*)&lds[r * Hsz + k];  // LDS broadcast
                acc[r] = fmaf(x.x, w0, fmaf(x.y, w1, fmaf(x.z, w2, fmaf(x.w, w3, acc[r]))));
            }
        }

        if (tid < 128) {
            // RoPE: d = column within q or k (0..63)
            const int   d    = tid & 63;
            const float freq = __powf(10000.0f, -(float)(2 * (d >> 1)) * (1.0f / 64.0f));
            const int   pos0 = rowbase & (Lsz - 1);
            _Float16* dst = ((tid < 64) ? qws : kws) + (size_t)rowbase * Dsz + d;
            #pragma unroll
            for (int r = 0; r < 16; ++r) {
                const float partner = __shfl_xor(acc[r], 1, 32);  // x[d^1], same wave
                const float x2  = (d & 1) ? partner : -partner;
                float s, c;
                __sincosf((float)(pos0 + r) * freq, &s, &c);
                dst[(size_t)r * Dsz] = (_Float16)(acc[r] * c + x2 * s);
            }
        } else {
            const int j = tid - 128;          // 0..23
            const int h = j >> 1;             // head
            const int w = j & 1;              // 0 = q-bias, 1 = k-bias
            const int bb   = rowbase >> 10;   // batch
            const int pos0 = rowbase & (Lsz - 1);
            float* plane = (w ? kbias : qbias) + (size_t)(bb * NHEADS + h) * Lsz + pos0;
            #pragma unroll
            for (int r = 0; r < 16; ++r) plane[r] = acc[r];
        }
    }
}

// ---------------------------------------------------------------------------
// Kernel 2: logits.  One wave32 per 16x16 (m,n) tile:
//   QK^T via 2x v_wmma_f32_16x16x32_f16 (K=64); transpose tile through LDS to
//   row-major float4 fragments; fan out over 12 heads with b128 NT stores
//   (output 402 MB > 192 MB L2 -> non-temporal).
// ---------------------------------------------------------------------------
__global__ void __launch_bounds__(256) qk_logits_kernel(
    const _Float16* __restrict__ qws, const _Float16* __restrict__ kws,
    const float* __restrict__ qbias, const float* __restrict__ kbias,
    const int* __restrict__ mask, float* __restrict__ out)
{
    __shared__ float tbuf[8][256];
    const int lane = threadIdx.x & 31;
    const int wv   = threadIdx.x >> 5;
    const int tile = blockIdx.x * 8 + wv;   // 32768 tiles
    const int nt = tile & 63;
    const int mt = (tile >> 6) & 63;
    const int b  = tile >> 12;
    const int m0 = mt << 4, n0 = nt << 4;
    const int hi = lane >> 4;      // lane group (0/1)
    const int lm = lane & 15;

    // A = qw (MxK), B = kw^T (KxN) -> contiguous k order, same for A and B:
    //   A lane group hi: elems 0..15 = K[16*hi .. 16*hi+15], M = lane%16
    //   B lane group hi: elems 0..15 = K[16*hi .. 16*hi+15], N = lane%16
    const _Float16* qrow = qws + ((size_t)(b * Lsz + m0 + lm) << 6);
    const _Float16* krow = kws + ((size_t)(b * Lsz + n0 + lm) << 6);
    const int ak0 = hi << 4;
    const int bk0 = hi << 4;

    const H8 q00 = *(const H8*)(qrow + ak0);
    const H8 q01 = *(const H8*)(qrow + ak0 + 8);
    const H8 q10 = *(const H8*)(qrow + ak0 + 32);
    const H8 q11 = *(const H8*)(qrow + ak0 + 40);
    const H8 k00 = *(const H8*)(krow + bk0);
    const H8 k01 = *(const H8*)(krow + bk0 + 8);
    const H8 k10 = *(const H8*)(krow + bk0 + 32);
    const H8 k11 = *(const H8*)(krow + bk0 + 40);

    v16h A0, A1, B0, B1;
    #pragma unroll
    for (int i = 0; i < 8; ++i) {
        A0[i] = q00.h[i]; A0[i + 8] = q01.h[i];
        A1[i] = q10.h[i]; A1[i + 8] = q11.h[i];
        B0[i] = k00.h[i]; B0[i + 8] = k01.h[i];
        B1[i] = k10.h[i]; B1[i + 8] = k11.h[i];
    }

    v8f c = {};
    c = __builtin_amdgcn_wmma_f32_16x16x32_f16(false, A0, false, B0, (short)0, c, false, false);
    c = __builtin_amdgcn_wmma_f32_16x16x32_f16(false, A1, false, B1, (short)0, c, false, false);

    // Transpose tile through LDS: WMMA layout (lane=column) -> row-major frags.
    // C/D layout: VGPR r -> row (8*hi + r), col lm.
    #pragma unroll
    for (int r = 0; r < 8; ++r)
        tbuf[wv][((hi << 3) + r) * 16 + lm] = c[r] * 0.125f;
    __syncthreads();   // one tile per wave -> single block-wide sync

    const int r0 = lane >> 2;        // 0..7: this lane's rows r0 and r0+8
    const int c0 = (lane & 3) << 2;  // cols c0..c0+3
    const f32x4 va = *(const f32x4*)&tbuf[wv][r0 * 16 + c0];
    const f32x4 vb = *(const f32x4*)&tbuf[wv][(r0 + 8) * 16 + c0];

    const int gm0 = m0 + r0;
    const int gm1 = gm0 + 8;
    const int gnb = n0 + c0;
    const int* mrow = mask + b * Lsz;
    const int mm0 = mrow[gm0];
    const int mm1 = mrow[gm1];
    const i32x4 mn = *(const i32x4*)&mrow[gnb];

    // Fold causal -1e12 into base values; precompute per-element dead flags.
    float a0[4], a1[4];
    bool  d0[4], d1[4];
    #pragma unroll
    for (int j = 0; j < 4; ++j) {
        const int gn = gnb + j;
        d0[j] = !(mm0 && mn[j]);
        d1[j] = !(mm1 && mn[j]);
        a0[j] = va[j] - ((gm0 > gn) ? 1.0e12f : 0.0f);
        a1[j] = vb[j] - ((gm1 > gn) ? 1.0e12f : 0.0f);
    }

    const float* qpl = qbias + (size_t)(b * NHEADS) * Lsz;
    const float* kpl = kbias + (size_t)(b * NHEADS) * Lsz;
    float* obase = out + ((size_t)(b * NHEADS) << 20) + gnb;

    for (int h = 0; h < NHEADS; ++h) {
        const float qb0 = qpl[h * Lsz + gm0];
        const float qb1 = qpl[h * Lsz + gm1];
        const f32x4 kb  = *(const f32x4*)&kpl[h * Lsz + gnb];
        f32x4 o0, o1;
        #pragma unroll
        for (int j = 0; j < 4; ++j) {
            o0[j] = d0[j] ? -__builtin_inff() : (a0[j] + qb0 + kb[j]);
            o1[j] = d1[j] ? -__builtin_inff() : (a1[j] + qb1 + kb[j]);
        }
        float* op = obase + ((size_t)h << 20);
        __builtin_nontemporal_store(o0, (f32x4*)(op + ((size_t)gm0 << 10)));
        __builtin_nontemporal_store(o1, (f32x4*)(op + ((size_t)gm1 << 10)));
    }
}

// ---------------------------------------------------------------------------
extern "C" void kernel_launch(void* const* d_in, const int* in_sizes, int n_in,
                              void* d_out, int out_size, void* d_ws, size_t ws_size,
                              hipStream_t stream) {
    const float* hidden = (const float*)d_in[0];
    const int*   mask   = (const int*)  d_in[1];
    const float* W1     = (const float*)d_in[2];
    const float* b1     = (const float*)d_in[3];
    const float* W2     = (const float*)d_in[4];
    const float* b2     = (const float*)d_in[5];
    float* out = (float*)d_out;

    // workspace layout (~3 MB total)
    char* ws = (char*)d_ws;
    _Float16* qws   = (_Float16*)(ws);                 // 8192*64 f16 = 1 MB
    _Float16* kws   = (_Float16*)(ws + 1048576);       // 1 MB
    float*    qbias = (float*)   (ws + 2097152);       // 8*12*1024 f32 = 384 KB
    float*    kbias = (float*)   (ws + 2490368);       // 384 KB
    float*    W12   = (float*)   (ws + 2883584);       // 768*24 f32 = 72 KB
    float*    bfull = (float*)   (ws + 2957312);       // 24 f32

    hipLaunchKernelGGL(w12_kernel, dim3(72), dim3(256), 0, stream,
                       W1, b1, W2, b2, W12, bfull);
    hipLaunchKernelGGL(seq_rope_kernel, dim3((Bsz * Lsz) / 16), dim3(256), 0, stream,
                       hidden, W1, b1, W12, bfull, qws, kws, qbias, kbias);
    hipLaunchKernelGGL(qk_logits_kernel, dim3((Bsz * 64 * 64) / 8), dim3(256), 0, stream,
                       qws, kws, qbias, kbias, mask, out);
}
